// MultiHeadAttention_33767032881412
// MI455X (gfx1250) — matrix operand order repaired and checked by reading, not compiled, not run on hardware
//
#include <hip/hip_runtime.h>
#include <hip/hip_bf16.h>

typedef __attribute__((ext_vector_type(16))) __bf16 v16bf;
typedef __attribute__((ext_vector_type(8)))  __bf16 v8bf;
typedef __attribute__((ext_vector_type(8)))  float  v8f;
typedef __attribute__((ext_vector_type(4)))  int    v4i;

#define HDIM 768
#define SEQ  1024
#define BATCH 8
#define NH   12
#define DH   64

// ---- async global->LDS staging (CDNA5), with safe fallback ----
#if defined(__has_builtin)
#  if __has_builtin(__builtin_amdgcn_global_load_async_to_lds_b128)
#    define HAVE_ASYNC_LDS 1
#  endif
#endif
#ifndef HAVE_ASYNC_LDS
#  define HAVE_ASYNC_LDS 0
#endif

// Builtin signature (from hipcc diagnostic): param0 = v4i __device__* (AS1 global
// source), param1 = v4i __shared__* (AS3 LDS dest), then imm offset, imm cpol.
typedef __attribute__((address_space(1))) v4i gv4i;
typedef __attribute__((address_space(3))) v4i lv4i;

// Copy 16 bytes (8 bf16) per lane from global to LDS.
__device__ __forceinline__ void stage16(const __bf16* g, __bf16* l) {
#if HAVE_ASYNC_LDS
    // flat->AS1 is value-identical for global memory; flat LDS addr low 32 bits
    // are the LDS offset (ISA aperture rules), so integer casts are exact.
    __builtin_amdgcn_global_load_async_to_lds_b128(
        (gv4i*)(unsigned long long)(size_t)g,
        (lv4i*)(unsigned int)(size_t)l, 0, 0);
#else
    *(v8bf*)l = *(const v8bf*)g;
#endif
}

__device__ __forceinline__ void stage_wait() {
#if HAVE_ASYNC_LDS
#  if __has_builtin(__builtin_amdgcn_s_wait_asynccnt)
    __builtin_amdgcn_s_wait_asynccnt(0);
#  else
    asm volatile("s_wait_asynccnt 0x0" ::: "memory");
#  endif
#endif
}

// Build a 16-element bf16 fragment from two contiguous 8-element LDS chunks.
__device__ __forceinline__ v16bf load_frag16(const __bf16* p0, const __bf16* p1) {
    v8bf lo = *(const v8bf*)p0;
    v8bf hi = *(const v8bf*)p1;
    v16bf r;
#pragma unroll
    for (int i = 0; i < 8; ++i) { r[i] = lo[i]; r[i + 8] = hi[i]; }
    return r;
}

// ---------------- prep kernels ----------------
__global__ void mha_cast_x(const float* __restrict__ x, __bf16* __restrict__ xb, int n) {
    for (int i = blockIdx.x * blockDim.x + threadIdx.x; i < n; i += gridDim.x * blockDim.x)
        xb[i] = (__bf16)x[i];
}

// wt[w][n][k] = W_w[k][n]  (transposed bf16 weights, 3 concatenated)
__global__ void mha_transpose_w(const float* __restrict__ Wq, const float* __restrict__ Wk,
                                const float* __restrict__ Wv, __bf16* __restrict__ wt) {
    const int total = 3 * HDIM * HDIM;
    for (int i = blockIdx.x * blockDim.x + threadIdx.x; i < total; i += gridDim.x * blockDim.x) {
        int w = i / (HDIM * HDIM);
        int r = i - w * (HDIM * HDIM);
        int n = r / HDIM, k = r - n * HDIM;
        const float* W = (w == 0) ? Wq : (w == 1) ? Wk : Wv;
        wt[i] = (__bf16)W[(size_t)k * HDIM + n];
    }
}

// ---------------- fused QKV projection GEMM ----------------
// grid (64, 18): 128-row tiles over 8192 rows, 128-col tiles over 3*768 cols.
// 8 waves, each owns a 16x128 strip: 1 A-frag + 8 B-frags -> 8 WMMAs per k-step.
__global__ __launch_bounds__(256) void mha_qkv_gemm(
    const __bf16* __restrict__ xb, const __bf16* __restrict__ wt,
    const float* __restrict__ bq, const float* __restrict__ bk, const float* __restrict__ bv,
    __bf16* __restrict__ Q, __bf16* __restrict__ K, __bf16* __restrict__ V) {
    __shared__ __bf16 As[128 * 40];   // 128 rows x 32 k, stride 40
    __shared__ __bf16 Bs[128 * 40];   // 128 n-rows x 32 k, stride 40

    const int tid = threadIdx.x;
    const int wave = tid >> 5, lane = tid & 31;
    const int hl = lane & 15, sel = lane >> 4;

    const int row0 = blockIdx.x * 128;
    const int n0 = blockIdx.y * 128;
    const int w = n0 / HDIM;                       // tiles never straddle a W boundary
    const int nl0 = n0 - w * HDIM;
    const __bf16* wtw = wt + (size_t)w * HDIM * HDIM;
    const float* bias = (w == 0) ? bq : (w == 1) ? bk : bv;

    v8f acc[8] = {};

    for (int k0 = 0; k0 < HDIM; k0 += 32) {
#pragma unroll
        for (int it = 0; it < 2; ++it) {   // A,B tiles: 128x32 each, 2 v8bf per thread
            int idx = (tid + it * 256) * 8;
            int r = idx >> 5, c = idx & 31;
            stage16(&xb[(size_t)(row0 + r) * HDIM + k0 + c], &As[r * 40 + c]);
            stage16(&wtw[(size_t)(nl0 + r) * HDIM + k0 + c], &Bs[r * 40 + c]);
        }
        stage_wait();
        __syncthreads();

        const int arow = wave * 16 + hl;
        const int akb = sel ? 8 : 0;               // A: K 0-7/16-23 vs 8-15/24-31
        v16bf a = load_frag16(&As[arow * 40 + akb], &As[arow * 40 + akb + 16]);
        const int bkb = sel ? 8 : 0;               // B: K 0-7/16-23 vs 8-15/24-31
#pragma unroll
        for (int f = 0; f < 8; ++f) {
            int nc = f * 16 + hl;
            v16bf b = load_frag16(&Bs[nc * 40 + bkb], &Bs[nc * 40 + bkb + 16]);
            acc[f] = __builtin_amdgcn_wmma_f32_16x16x32_bf16(false, a, false, b,
                                                             (short)0, acc[f], false, false);
        }
        __syncthreads();
    }

    // epilogue: bias + scatter to [b, h, s, d] bf16
    const int bidx = row0 >> 10;                   // 128 | 1024, constant per block
    __bf16* dst = (w == 0) ? Q : (w == 1) ? K : V;
#pragma unroll
    for (int f = 0; f < 8; ++f) {
        int col = nl0 + f * 16 + hl;               // 0..767 within this W
        int h = col >> 6, d = col & 63;
        float bi = bias[col];
#pragma unroll
        for (int v = 0; v < 8; ++v) {
            int m = sel * 8 + v;
            int s = (row0 + wave * 16 + m) & (SEQ - 1);
            dst[(size_t)((bidx * NH + h) * SEQ + s) * DH + d] = (__bf16)(acc[f][v] + bi);
        }
    }
}

// ---------------- flash attention ----------------
// grid (8, 96): 128-query tile x (b*12+h)
__global__ __launch_bounds__(256) void mha_flash_attn(
    const __bf16* __restrict__ Q, const __bf16* __restrict__ K,
    const __bf16* __restrict__ V, float* __restrict__ ctx) {
    __shared__ __bf16 Qs[128 * 72];   // [q][d]
    __shared__ __bf16 Ks[64 * 72];    // [k][d] (== B-transposed layout for Q*K^T)
    __shared__ __bf16 Vt[64 * 72];    // [d][k] (== B-transposed layout for P*V)
    __shared__ __bf16 Ps[128 * 72];   // [q][k], per-wave 16-row slices

    const int tid = threadIdx.x;
    const int wave = tid >> 5, lane = tid & 31;
    const int hl = lane & 15, sel = lane >> 4;
    const int bh = blockIdx.y;
    const int q0 = blockIdx.x * 128;
    const size_t base = (size_t)bh * SEQ * DH;

#pragma unroll
    for (int it = 0; it < 4; ++it) {  // Q tile 128x64, async staged
        int idx = (tid + it * 256) * 8;
        int r = idx >> 6, c = idx & 63;
        stage16(&Q[base + (size_t)(q0 + r) * DH + c], &Qs[r * 72 + c]);
    }

    float mrow[8], lrow[8];
    v8f o[4] = {};
#pragma unroll
    for (int v = 0; v < 8; ++v) { mrow[v] = -3.0e38f; lrow[v] = 0.0f; }

    const int qr = wave * 16 + hl;    // A-fragment row for this wave's slice

    for (int kb0 = 0; kb0 < SEQ; kb0 += 64) {
        __syncthreads();              // Ks/Vt reads from previous iteration done
#pragma unroll
        for (int it = 0; it < 2; ++it) {   // K tile 64x64 -> Ks[k][d], async staged
            int idx = (tid + it * 256) * 8;
            int r = idx >> 6, c = idx & 63;
            stage16(&K[base + (size_t)(kb0 + r) * DH + c], &Ks[r * 72 + c]);
        }
        {   // V tile transposed -> Vt[d][k] (VGPR path: needs in-flight transpose)
            int r = tid >> 2;
            int c0 = (tid & 3) * 16;
            v8bf v0 = *(const v8bf*)&V[base + (size_t)(kb0 + r) * DH + c0];
            v8bf v1 = *(const v8bf*)&V[base + (size_t)(kb0 + r) * DH + c0 + 8];
#pragma unroll
            for (int i = 0; i < 8; ++i) {
                Vt[(c0 + i) * 72 + r] = v0[i];
                Vt[(c0 + 8 + i) * 72 + r] = v1[i];
            }
        }
        if (kb0 + 64 < SEQ)   // prefetch next V tile -> global_prefetch_b8
            __builtin_prefetch(&V[base + (size_t)(kb0 + 64 + (tid >> 2)) * DH], 0, 0);
        stage_wait();                 // all async copies (incl. Q on first pass) in LDS
        __syncthreads();

        // S = Q*K^T (16 q-rows x 64 keys per wave)
        v8f s[4] = {};
#pragma unroll
        for (int ks = 0; ks < 2; ++ks) {
            int akb = ks * 32 + (sel ? 8 : 0);
            v16bf a = load_frag16(&Qs[qr * 72 + akb], &Qs[qr * 72 + akb + 16]);
            int bkb = ks * 32 + (sel ? 8 : 0);
#pragma unroll
            for (int f = 0; f < 4; ++f) {
                int nc = f * 16 + hl;
                v16bf b = load_frag16(&Ks[nc * 72 + bkb], &Ks[nc * 72 + bkb + 16]);
                s[f] = __builtin_amdgcn_wmma_f32_16x16x32_bf16(false, a, false, b,
                                                               (short)0, s[f], false, false);
            }
        }

        // online softmax over this key block
#pragma unroll
        for (int v = 0; v < 8; ++v) {
            float bm = -3.0e38f;
#pragma unroll
            for (int f = 0; f < 4; ++f) { float e = s[f][v] * 0.125f; s[f][v] = e; bm = fmaxf(bm, e); }
            bm = fmaxf(bm, __shfl_xor(bm, 1, 32));
            bm = fmaxf(bm, __shfl_xor(bm, 2, 32));
            bm = fmaxf(bm, __shfl_xor(bm, 4, 32));
            bm = fmaxf(bm, __shfl_xor(bm, 8, 32));
            float mn = fmaxf(mrow[v], bm);
            float alpha = __expf(mrow[v] - mn);
            float rs = 0.0f;
#pragma unroll
            for (int f = 0; f < 4; ++f) { float p = __expf(s[f][v] - mn); s[f][v] = p; rs += p; }
            rs += __shfl_xor(rs, 1, 32);
            rs += __shfl_xor(rs, 2, 32);
            rs += __shfl_xor(rs, 4, 32);
            rs += __shfl_xor(rs, 8, 32);
            lrow[v] = lrow[v] * alpha + rs;
            mrow[v] = mn;
#pragma unroll
            for (int f = 0; f < 4; ++f) o[f][v] *= alpha;
            int prow = wave * 16 + sel * 8 + v;     // C-layout -> A-layout via per-wave LDS slice
#pragma unroll
            for (int f = 0; f < 4; ++f) Ps[prow * 72 + f * 16 + hl] = (__bf16)s[f][v];
        }

        // O += P*V (same-wave LDS producer/consumer: DS ops are in-order per wave)
#pragma unroll
        for (int ks = 0; ks < 2; ++ks) {
            int akb = ks * 32 + (sel ? 8 : 0);
            v16bf a = load_frag16(&Ps[qr * 72 + akb], &Ps[qr * 72 + akb + 16]);
            int bkb = ks * 32 + (sel ? 8 : 0);
#pragma unroll
            for (int f = 0; f < 4; ++f) {
                int nc = f * 16 + hl;
                v16bf b = load_frag16(&Vt[nc * 72 + bkb], &Vt[nc * 72 + bkb + 16]);
                o[f] = __builtin_amdgcn_wmma_f32_16x16x32_bf16(false, a, false, b,
                                                               (short)0, o[f], false, false);
            }
        }
    }

    // normalize and scatter to ctx[b, s, h*64+d] (fp32)
    const int b = bh / NH, h = bh % NH;
#pragma unroll
    for (int f = 0; f < 4; ++f) {
        int d = f * 16 + hl;
#pragma unroll
        for (int v = 0; v < 8; ++v) {
            int m = sel * 8 + v;
            int q = q0 + wave * 16 + m;
            ctx[((size_t)(b * SEQ + q)) * HDIM + h * DH + d] = o[f][v] / lrow[v];
        }
    }
}

// ---------------- residual + layernorm ----------------
__global__ __launch_bounds__(256) void mha_residual_ln(
    const float* __restrict__ x, const float* __restrict__ ctx,
    const float* __restrict__ gamma, const float* __restrict__ beta,
    float* __restrict__ out) {
    __shared__ float ws1[8], ws2[8];
    const int row = blockIdx.x;
    const float* xr = x + (size_t)row * HDIM;
    const float* cr = ctx + (size_t)row * HDIM;

    float y[3];
    float sum = 0.0f;
#pragma unroll
    for (int i = 0; i < 3; ++i) {
        int c = threadIdx.x + i * 256;
        y[i] = xr[c] + 2.0f * cr[c];          // DropPath identity: x + 2*context
        sum += y[i];
    }
#pragma unroll
    for (int m = 1; m < 32; m <<= 1) sum += __shfl_xor(sum, m, 32);
    const int wv = threadIdx.x >> 5, ln = threadIdx.x & 31;
    if (ln == 0) ws1[wv] = sum;
    __syncthreads();
    float tot = 0.0f;
#pragma unroll
    for (int i = 0; i < 8; ++i) tot += ws1[i];
    const float mu = tot * (1.0f / (float)HDIM);

    float vs = 0.0f;
#pragma unroll
    for (int i = 0; i < 3; ++i) { float d = y[i] - mu; vs += d * d; }
#pragma unroll
    for (int m = 1; m < 32; m <<= 1) vs += __shfl_xor(vs, m, 32);
    if (ln == 0) ws2[wv] = vs;
    __syncthreads();
    float vtot = 0.0f;
#pragma unroll
    for (int i = 0; i < 8; ++i) vtot += ws2[i];
    const float rstd = rsqrtf(vtot * (1.0f / (float)HDIM) + 1e-6f);

#pragma unroll
    for (int i = 0; i < 3; ++i) {
        int c = threadIdx.x + i * 256;
        out[(size_t)row * HDIM + c] = (y[i] - mu) * rstd * gamma[c] + beta[c];
    }
}

extern "C" void kernel_launch(void* const* d_in, const int* in_sizes, int n_in,
                              void* d_out, int out_size, void* d_ws, size_t ws_size,
                              hipStream_t stream) {
    (void)in_sizes; (void)n_in; (void)out_size; (void)ws_size;
    const float* x  = (const float*)d_in[0];
    const float* Wq = (const float*)d_in[1];
    const float* bq = (const float*)d_in[2];
    const float* Wk = (const float*)d_in[3];
    const float* bk = (const float*)d_in[4];
    const float* Wv = (const float*)d_in[5];
    const float* bv = (const float*)d_in[6];
    const float* g  = (const float*)d_in[7];
    const float* be = (const float*)d_in[8];
    float* out = (float*)d_out;

    const size_t NX = (size_t)BATCH * SEQ * HDIM;   // 6,291,456
    char* p = (char*)d_ws;
    auto alloc = [&](size_t bytes) { char* r = p; p += (bytes + 255) & ~(size_t)255; return r; };
    __bf16* xb  = (__bf16*)alloc(NX * 2);
    __bf16* wt  = (__bf16*)alloc((size_t)3 * HDIM * HDIM * 2);
    __bf16* Qb  = (__bf16*)alloc(NX * 2);
    __bf16* Kb  = (__bf16*)alloc(NX * 2);
    __bf16* Vb  = (__bf16*)alloc(NX * 2);
    float*  ctx = (float*)alloc(NX * 4);

    mha_cast_x<<<4096, 256, 0, stream>>>(x, xb, (int)NX);
    mha_transpose_w<<<4096, 256, 0, stream>>>(Wq, Wk, Wv, wt);
    mha_qkv_gemm<<<dim3(64, 18), 256, 0, stream>>>(xb, wt, bq, bk, bv, Qb, Kb, Vb);
    mha_flash_attn<<<dim3(8, 96), 256, 0, stream>>>(Qb, Kb, Vb, ctx);
    mha_residual_ln<<<BATCH * SEQ, 256, 0, stream>>>(x, ctx, g, be, out);
}
